// TorchGatherEinsum_24902220382295
// MI455X (gfx1250) — matrix operand order repaired and checked by reading, not compiled, not run on hardware
//
#include <hip/hip_runtime.h>

typedef __attribute__((ext_vector_type(16))) __bf16         bf16x16;
typedef __attribute__((ext_vector_type(8)))  float          f32x8;
typedef __attribute__((ext_vector_type(4)))  float          f32x4;
typedef __attribute__((ext_vector_type(8)))  unsigned short u16x8;
typedef __attribute__((ext_vector_type(4)))  unsigned short u16x4;
typedef __attribute__((ext_vector_type(4)))  int            i32x4;

#define B_ 8
#define T_ 2048
#define I_ 512
#define E_ 16
#define K_ 256
#define J_ 512

#define TM 128       // output rows per workgroup
#define TN 128       // output cols per workgroup
#define TK 64        // I-chunk per stage
#define NSTAGE (I_ / TK)
#define PITCH 72     // LDS K-pitch in bf16 units (144B, 16B-aligned, breaks bank stride)

// ---- feature probes (device pass only) ----
#if defined(__HIP_DEVICE_COMPILE__)
#  if __has_builtin(__builtin_amdgcn_global_load_async_to_lds_b128)
#    define HAVE_ASYNC 1
#  endif
#  if __has_builtin(__builtin_amdgcn_s_wait_asynccnt)
#    define HAVE_WAITASYNC 1
#  endif
#endif
#ifndef HAVE_ASYNC
#  define HAVE_ASYNC 0
#endif
#ifndef HAVE_WAITASYNC
#  define HAVE_WAITASYNC 0
#endif

#if HAVE_ASYNC
typedef __attribute__((address_space(1))) i32x4* gas_b128;
typedef __attribute__((address_space(3))) i32x4* lds_b128;
#endif

__device__ __forceinline__ unsigned short f2bf_rne(float x) {
    unsigned u = __float_as_uint(x);
    return (unsigned short)((u + 0x7FFFu + ((u >> 16) & 1u)) >> 16);
}

union Frag {
    bf16x16 v;
    u16x8   h[2];
};

// 16-byte global -> LDS copy (async DMA path when available)
__device__ __forceinline__ void cp16(unsigned short* l, const unsigned short* g) {
#if HAVE_ASYNC
    __builtin_amdgcn_global_load_async_to_lds_b128(
        (gas_b128)(unsigned short*)g, (lds_b128)l, 0, 0);
#else
    *(u16x8*)l = *(const u16x8*)g;
#endif
}

__device__ __forceinline__ void wait_async_le16() {
#if HAVE_ASYNC
#  if HAVE_WAITASYNC
    __builtin_amdgcn_s_wait_asynccnt(16);
#  else
    asm volatile("s_wait_asynccnt 0x10" ::: "memory");
#  endif
#endif
}

__device__ __forceinline__ void wait_async_0() {
#if HAVE_ASYNC
#  if HAVE_WAITASYNC
    __builtin_amdgcn_s_wait_asynccnt(0);
#  else
    asm volatile("s_wait_asynccnt 0x0" ::: "memory");
#  endif
#endif
}

// ------------------------------------------------------------------
// Prepass 1: X[b,t,i] fp32 -> Xhi/Xlo bf16 (RNE hi, residual lo)
// ------------------------------------------------------------------
__global__ __launch_bounds__(256)
void convert_x(const float* __restrict__ X,
               unsigned short* __restrict__ xhi,
               unsigned short* __restrict__ xlo)
{
    size_t t = (size_t)blockIdx.x * 256 + threadIdx.x;   // float4 id
    f32x4 x = ((const f32x4*)X)[t];
    u16x4 hi, lo;
#pragma unroll
    for (int j = 0; j < 4; ++j) {
        unsigned short hb = f2bf_rne(x[j]);
        float hf = __uint_as_float((unsigned)hb << 16);
        hi[j] = hb;
        lo[j] = f2bf_rne(x[j] - hf);
    }
    ((u16x4*)xhi)[t] = hi;
    ((u16x4*)xlo)[t] = lo;
}

// ------------------------------------------------------------------
// Prepass 2: W[e,i,j] fp32 -> Whi/Wlo bf16 transposed to [e,j,i]
// (tiled 64x64 transpose through LDS)
// ------------------------------------------------------------------
__global__ __launch_bounds__(256)
void convert_w(const float* __restrict__ W,
               unsigned short* __restrict__ whi,
               unsigned short* __restrict__ wlo)
{
    __shared__ unsigned short th[64][68];
    __shared__ unsigned short tl[64][68];

    const int tid = threadIdx.x;
    const int bid = blockIdx.x;           // e*(8*8) + ti*8 + tj
    const int tj  = bid & 7;
    const int ti  = (bid >> 3) & 7;
    const int e   = bid >> 6;
    const int i0  = ti * 64;
    const int j0  = tj * 64;

    const float* Wg = W + (size_t)e * I_ * J_;
#pragma unroll
    for (int p = 0; p < 4; ++p) {
        int u   = tid + p * 256;          // float4 id within 64x64 tile
        int row = u >> 4;                 // i within tile
        int c4  = u & 15;
        f32x4 x = *(const f32x4*)(Wg + (size_t)(i0 + row) * J_ + j0 + c4 * 4);
#pragma unroll
        for (int j = 0; j < 4; ++j) {
            unsigned short hb = f2bf_rne(x[j]);
            float hf = __uint_as_float((unsigned)hb << 16);
            th[row][c4 * 4 + j] = hb;
            tl[row][c4 * 4 + j] = f2bf_rne(x[j] - hf);
        }
    }
    __syncthreads();

    unsigned short* Whe = whi + (size_t)e * J_ * I_;
    unsigned short* Wle = wlo + (size_t)e * J_ * I_;
#pragma unroll
    for (int p = 0; p < 4; ++p) {
        int u  = tid + p * 256;
        int j  = u >> 4;                  // j within tile
        int i4 = u & 15;                  // u16x4 block along i
        u16x4 h, l;
#pragma unroll
        for (int q = 0; q < 4; ++q) {
            h[q] = th[i4 * 4 + q][j];
            l[q] = tl[i4 * 4 + q][j];
        }
        *(u16x4*)(Whe + (size_t)(j0 + j) * I_ + i0 + i4 * 4) = h;
        *(u16x4*)(Wle + (size_t)(j0 + j) * I_ + i0 + i4 * 4) = l;
    }
}

// ------------------------------------------------------------------
// Fast main kernel: bf16 operands preconverted; async double-buffered
// LDS staging; bf16x3 split-product WMMA with fp32 accumulate.
// ------------------------------------------------------------------
__global__ __launch_bounds__(256)
void gemm_fast(const unsigned short* __restrict__ Xhi,
               const unsigned short* __restrict__ Xlo,
               const unsigned short* __restrict__ Whi,
               const unsigned short* __restrict__ Wlo,
               const long long* __restrict__ ind,
               float* __restrict__ Y)
{
    __shared__ unsigned short sAhi[2][TM * PITCH];
    __shared__ unsigned short sAlo[2][TM * PITCH];
    __shared__ unsigned short sBhi[2][TN * PITCH];
    __shared__ unsigned short sBlo[2][TN * PITCH];

    const int tid  = threadIdx.x;
    const int lane = tid & 31;
    const int wvid = tid >> 5;
    const int wm   = wvid >> 2;
    const int wn   = wvid & 3;
    const int half = lane >> 4;
    const int lm   = lane & 15;

    const int bid = blockIdx.x;
    const int tn  = bid & 3;
    const int tm  = (bid >> 2) & 1;
    const int e   = (bid >> 3) & 15;
    const int b   = bid >> 7;

    const int m0 = tm * TM;
    const int n0 = tn * TN;

    const long long* indp = ind + ((size_t)(b * E_ + e)) * K_ + m0;

    // A: 128 rows x 64 bf16 per stage = 1024 u16x8 chunks; 4 per thread
    const unsigned short* ahip[4];
    const unsigned short* alop[4];
    int aoffL[4];
#pragma unroll
    for (int i = 0; i < 4; ++i) {
        int u   = tid + i * 256;
        int row = u >> 3;
        int c8  = u & 7;
        long long idx = indp[row];
        size_t base = ((size_t)b * T_ + (size_t)idx) * I_ + c8 * 8;
        ahip[i]  = Xhi + base;
        alop[i]  = Xlo + base;
        aoffL[i] = row * PITCH + c8 * 8;
    }

    // B: Wt[e][j][i], chunk = 128 j x 64 i bf16; 4 u16x8 per thread
    const unsigned short* whp = Whi + (size_t)e * J_ * I_ + (size_t)n0 * I_;
    const unsigned short* wlp = Wlo + (size_t)e * J_ * I_ + (size_t)n0 * I_;
    int woffG[4], woffL[4];
#pragma unroll
    for (int i = 0; i < 4; ++i) {
        int u  = tid + i * 256;
        int j  = u >> 3;
        int c8 = u & 7;
        woffG[i] = j * I_ + c8 * 8;
        woffL[i] = j * PITCH + c8 * 8;
    }

    auto issue = [&](int s, int buf) {
#pragma unroll
        for (int i = 0; i < 4; ++i) {
            cp16(&sAhi[buf][aoffL[i]], ahip[i] + s * TK);
            cp16(&sAlo[buf][aoffL[i]], alop[i] + s * TK);
        }
#pragma unroll
        for (int i = 0; i < 4; ++i) {
            cp16(&sBhi[buf][woffL[i]], whp + woffG[i] + s * TK);
            cp16(&sBlo[buf][woffL[i]], wlp + woffG[i] + s * TK);
        }
    };

    f32x8 acc[4][2] = {};

    issue(0, 0);
    for (int s = 0; s < NSTAGE; ++s) {
        const int cur = s & 1;
        if (s + 1 < NSTAGE) {
            issue(s + 1, 1 - cur);   // prefetch next stage into other buffer
            wait_async_le16();       // retire current stage's 16 async copies
        } else {
            wait_async_0();
        }
        __syncthreads();

        const unsigned short* A_hi = sAhi[cur];
        const unsigned short* A_lo = sAlo[cur];
        const unsigned short* B_hi = sBhi[cur];
        const unsigned short* B_lo = sBlo[cur];

#pragma unroll
        for (int kk = 0; kk < TK; kk += 32) {
            // Hoist ALL fragment loads for this k-step into distinct registers
            // so the 24 ds_load_b128 issue up-front and the 24 WMMAs run
            // back-to-back without full dscnt stalls from register reuse.
            Frag bh[2], bl[2];
#pragma unroll
            for (int nt = 0; nt < 2; ++nt) {
                int ncol = wn * 32 + nt * 16 + lm;
                const u16x8* ph = (const u16x8*)&B_hi[ncol * PITCH + kk + half * 16];
                bh[nt].h[0] = ph[0];
                bh[nt].h[1] = ph[1];
                const u16x8* pl = (const u16x8*)&B_lo[ncol * PITCH + kk + half * 16];
                bl[nt].h[0] = pl[0];
                bl[nt].h[1] = pl[1];
            }
            Frag ah[4], al[4];
#pragma unroll
            for (int mt = 0; mt < 4; ++mt) {
                int arow = wm * 64 + mt * 16 + lm;
                int o1 = arow * PITCH + kk + half * 16;
                ah[mt].h[0] = *(const u16x8*)&A_hi[o1];
                ah[mt].h[1] = *(const u16x8*)&A_hi[o1 + 8];
                al[mt].h[0] = *(const u16x8*)&A_lo[o1];
                al[mt].h[1] = *(const u16x8*)&A_lo[o1 + 8];
            }
#pragma unroll
            for (int mt = 0; mt < 4; ++mt) {
#pragma unroll
                for (int nt = 0; nt < 2; ++nt) {
                    acc[mt][nt] = __builtin_amdgcn_wmma_f32_16x16x32_bf16(
                        false, ah[mt].v, false, bh[nt].v, (short)0, acc[mt][nt], false, false);
                    acc[mt][nt] = __builtin_amdgcn_wmma_f32_16x16x32_bf16(
                        false, ah[mt].v, false, bl[nt].v, (short)0, acc[mt][nt], false, false);
                    acc[mt][nt] = __builtin_amdgcn_wmma_f32_16x16x32_bf16(
                        false, al[mt].v, false, bh[nt].v, (short)0, acc[mt][nt], false, false);
                }
            }
        }
        __syncthreads();
    }

    float* Yp = Y + (((size_t)(b * E_ + e)) * K_ + m0) * J_ + n0;
#pragma unroll
    for (int mt = 0; mt < 4; ++mt) {
        int rbase = wm * 64 + mt * 16 + half * 8;
#pragma unroll
        for (int nt = 0; nt < 2; ++nt) {
            int col = wn * 32 + nt * 16 + lm;
#pragma unroll
            for (int v = 0; v < 8; ++v) {
                Yp[(size_t)(rbase + v) * J_ + col] = acc[mt][nt][v];
            }
        }
    }
}

// ------------------------------------------------------------------
// Fallback (ws too small): round-1 kernel, truncate-hi split (cheap).
// ------------------------------------------------------------------
__global__ __launch_bounds__(256)
void gemm_fallback(const float* __restrict__ X,
                   const long long* __restrict__ ind,
                   const float* __restrict__ W,
                   float* __restrict__ Y)
{
    __shared__ unsigned short sAhi[TM * PITCH];
    __shared__ unsigned short sAlo[TM * PITCH];
    __shared__ unsigned short sBhi[TN * PITCH];
    __shared__ unsigned short sBlo[TN * PITCH];

    const int tid  = threadIdx.x;
    const int lane = tid & 31;
    const int wvid = tid >> 5;
    const int wm   = wvid >> 2;
    const int wn   = wvid & 3;
    const int half = lane >> 4;
    const int lm   = lane & 15;

    const int bid = blockIdx.x;
    const int tn  = bid & 3;
    const int tm  = (bid >> 2) & 1;
    const int e   = (bid >> 3) & 15;
    const int b   = bid >> 7;

    const int m0 = tm * TM;
    const int n0 = tn * TN;

    const float*     Xb   = X + (size_t)b * T_ * I_;
    const float*     Wg   = W + (size_t)e * I_ * J_ + n0;
    const long long* indp = ind + ((size_t)(b * E_ + e)) * K_ + m0;

    const float* aptr[8];
    int          aoff[8];
#pragma unroll
    for (int i = 0; i < 8; ++i) {
        int u   = tid + i * 256;
        int row = u >> 4;
        int c4  = u & 15;
        long long idx = indp[row];
        aptr[i] = Xb + (size_t)idx * I_ + c4 * 4;
        aoff[i] = row * PITCH + c4 * 4;
    }

    f32x8 acc[4][2] = {};

    for (int s = 0; s < NSTAGE; ++s) {
#pragma unroll
        for (int i = 0; i < 8; ++i) {
            f32x4 x = *(const f32x4*)(aptr[i] + s * TK);
            u16x4 hi, lo;
#pragma unroll
            for (int j = 0; j < 4; ++j) {
                unsigned u = __float_as_uint(x[j]);
                hi[j] = (unsigned short)(u >> 16);
                float hf = __uint_as_float(u & 0xFFFF0000u);
                lo[j] = f2bf_rne(x[j] - hf);
            }
            *(u16x4*)&sAhi[aoff[i]] = hi;
            *(u16x4*)&sAlo[aoff[i]] = lo;
            if (s + 1 < NSTAGE)
                __builtin_prefetch(aptr[i] + (s + 1) * TK, 0, 0);
        }
#pragma unroll
        for (int i = 0; i < 8; ++i) {
            int u   = tid + i * 256;
            int row = u >> 5;
            int c4  = u & 31;
            const float* wp = Wg + (size_t)(s * TK + row) * J_ + c4 * 4;
            f32x4 x = *(const f32x4*)wp;
#pragma unroll
            for (int j = 0; j < 4; ++j) {
                int col = c4 * 4 + j;
                unsigned uj = __float_as_uint(x[j]);
                sBhi[col * PITCH + row] = (unsigned short)(uj >> 16);
                float hf = __uint_as_float(uj & 0xFFFF0000u);
                sBlo[col * PITCH + row] = f2bf_rne(x[j] - hf);
            }
            if (s + 1 < NSTAGE)
                __builtin_prefetch(wp + (size_t)TK * J_, 0, 0);
        }
        __syncthreads();

#pragma unroll
        for (int kk = 0; kk < TK; kk += 32) {
            Frag bh[2], bl[2];
#pragma unroll
            for (int nt = 0; nt < 2; ++nt) {
                int ncol = wn * 32 + nt * 16 + lm;
                const u16x8* ph = (const u16x8*)&sBhi[ncol * PITCH + kk + half * 16];
                bh[nt].h[0] = ph[0];
                bh[nt].h[1] = ph[1];
                const u16x8* pl = (const u16x8*)&sBlo[ncol * PITCH + kk + half * 16];
                bl[nt].h[0] = pl[0];
                bl[nt].h[1] = pl[1];
            }
#pragma unroll
            for (int mt = 0; mt < 4; ++mt) {
                Frag ah, al;
                int arow = wm * 64 + mt * 16 + lm;
                int o1 = arow * PITCH + kk + half * 16;
                ah.h[0] = *(const u16x8*)&sAhi[o1];
                ah.h[1] = *(const u16x8*)&sAhi[o1 + 8];
                al.h[0] = *(const u16x8*)&sAlo[o1];
                al.h[1] = *(const u16x8*)&sAlo[o1 + 8];
#pragma unroll
                for (int nt = 0; nt < 2; ++nt) {
                    acc[mt][nt] = __builtin_amdgcn_wmma_f32_16x16x32_bf16(
                        false, ah.v, false, bh[nt].v, (short)0, acc[mt][nt], false, false);
                    acc[mt][nt] = __builtin_amdgcn_wmma_f32_16x16x32_bf16(
                        false, ah.v, false, bl[nt].v, (short)0, acc[mt][nt], false, false);
                    acc[mt][nt] = __builtin_amdgcn_wmma_f32_16x16x32_bf16(
                        false, al.v, false, bh[nt].v, (short)0, acc[mt][nt], false, false);
                }
            }
        }
        __syncthreads();
    }

    float* Yp = Y + (((size_t)(b * E_ + e)) * K_ + m0) * J_ + n0;
#pragma unroll
    for (int mt = 0; mt < 4; ++mt) {
        int rbase = wm * 64 + mt * 16 + half * 8;
#pragma unroll
        for (int nt = 0; nt < 2; ++nt) {
            int col = wn * 32 + nt * 16 + lm;
#pragma unroll
            for (int v = 0; v < 8; ++v) {
                Yp[(size_t)(rbase + v) * J_ + col] = acc[mt][nt][v];
            }
        }
    }
}

extern "C" void kernel_launch(void* const* d_in, const int* in_sizes, int n_in,
                              void* d_out, int out_size, void* d_ws, size_t ws_size,
                              hipStream_t stream) {
    const float*     X   = (const float*)d_in[0];
    const long long* ind = (const long long*)d_in[1];
    const float*     W   = (const float*)d_in[2];
    float*           Y   = (float*)d_out;

    const size_t nX = (size_t)B_ * T_ * I_;      // 8,388,608 elements
    const size_t nW = (size_t)E_ * I_ * J_;      // 4,194,304 elements
    const size_t need = (2 * nX + 2 * nW) * sizeof(unsigned short);  // 50,331,648 B

    dim3 block(256);
    dim3 grid(B_ * E_ * (K_ / TM) * (J_ / TN));  // 1024 workgroups

    if (ws_size >= need) {
        unsigned short* xhi = (unsigned short*)d_ws;
        unsigned short* xlo = xhi + nX;
        unsigned short* whi = xlo + nX;
        unsigned short* wlo = whi + nW;

        hipLaunchKernelGGL(convert_x, dim3((unsigned)(nX / 4 / 256)), block, 0, stream,
                           X, xhi, xlo);
        hipLaunchKernelGGL(convert_w, dim3(E_ * (I_ / 64) * (J_ / 64)), block, 0, stream,
                           W, whi, wlo);
        hipLaunchKernelGGL(gemm_fast, grid, block, 0, stream,
                           xhi, xlo, whi, wlo, ind, Y);
    } else {
        hipLaunchKernelGGL(gemm_fallback, grid, block, 0, stream, X, ind, W, Y);
    }
}
